// PolymorphicSNN_78623671321015
// MI455X (gfx1250) — matrix operand order repaired and checked by reading, not compiled, not run on hardware
//
#include <hip/hip_runtime.h>
#include <hip/hip_bf16.h>

// ---------------------------------------------------------------------------
// Shapes
#define BB   2048
#define NN   1024
#define KK   1024
#define PP   8
#define MM   8
// ---------------------------------------------------------------------------

typedef __attribute__((ext_vector_type(16))) __bf16 v16bf;
typedef __attribute__((ext_vector_type(8)))  float  v8f;

struct Frag { uint4 q[2]; };

__device__ __forceinline__ v16bf frag_cast(const Frag& f) {
    union { Frag f; v16bf v; } u;
    u.f = f;
    return u.v;
}

// ---------------- K1: fp32 -> bf16 (round-to-nearest-even) ------------------
__global__ __launch_bounds__(256) void cvt_bf16_kernel(const float* __restrict__ a,
                                                       unsigned short* __restrict__ o,
                                                       int n) {
    int i = blockIdx.x * blockDim.x + threadIdx.x;
    if (i < n) {
        unsigned int u = __float_as_uint(a[i]);
        unsigned int r = u + 0x7fffu + ((u >> 16) & 1u);
        o[i] = (unsigned short)(r >> 16);
    }
}

// ---------------- K2: xl = x @ W_lin^T + b_lin via v_wmma bf16 --------------
// NT GEMM: xl[b][n] = sum_k xb[b][k] * wb[n][k].  Both operands have K
// contiguous in memory, so A and B WMMA fragments are contiguous 16B loads.
// Wave computes a 16x64 strip (4 accumulators), K-steps of 32, with explicit
// double buffering so the L2 fragment loads for step kk+32 overlap the 4
// WMMAs of step kk (avoids s_wait_loadcnt 0 in front of every v_wmma).
__global__ __launch_bounds__(256) void gemm_xl_kernel(const unsigned short* __restrict__ xb,
                                                      const unsigned short* __restrict__ wb,
                                                      const float* __restrict__ b_lin,
                                                      float* __restrict__ xl) {
    const int wave = (blockIdx.x * blockDim.x + threadIdx.x) >> 5;
    const int lane = threadIdx.x & 31;
    const int rt = wave >> 4;          // 0..127  (row tile of 16)
    const int cs = wave & 15;          // 0..15   (64-col strip)
    const int r0 = rt * 16, c0 = cs * 64;

    const int rowA = r0 + (lane & 15);
    const int kA   = (lane < 16) ? 0 : 8;    // bf16 A-layout half-wave K offset
    const int colB = (lane & 15);
    const int kB   = (lane < 16) ? 0 : 8;    // bf16 B-layout half-wave K offset

    const unsigned short* aBase = xb + (size_t)rowA * KK + kA;
    const unsigned short* bBase = wb + (size_t)(c0 + colB) * KK + kB;

    v8f acc[4];
    #pragma unroll
    for (int s = 0; s < 4; ++s)
        #pragma unroll
        for (int r = 0; r < 8; ++r) acc[s][r] = 0.0f;

    auto loadA = [&](int kk) {
        Frag f;
        const uint4* ap = (const uint4*)(aBase + kk);
        f.q[0] = ap[0];     // K = kk+kA+0..7
        f.q[1] = ap[2];     // K = kk+kA+16..23
        return f;
    };
    auto loadB = [&](int kk, int s) {
        Frag f;
        const uint4* bp = (const uint4*)(bBase + (size_t)s * 16 * KK + kk);
        f.q[0] = bp[0];     // K = kk+kB+0..7
        f.q[1] = bp[2];     // K = kk+kB+16..23
        return f;
    };

    Frag aC = loadA(0);
    Frag bC[4];
    #pragma unroll
    for (int s = 0; s < 4; ++s) bC[s] = loadB(0, s);

    for (int kk = 0; kk < KK; kk += 32) {
        const int kn = (kk + 32 < KK) ? (kk + 32) : kk;   // dummy reload last iter
        Frag aN = loadA(kn);
        Frag bN[4];
        #pragma unroll
        for (int s = 0; s < 4; ++s) bN[s] = loadB(kn, s);

        const v16bf av = frag_cast(aC);
        #pragma unroll
        for (int s = 0; s < 4; ++s)
            acc[s] = __builtin_amdgcn_wmma_f32_16x16x32_bf16(
                false, av, false, frag_cast(bC[s]), (short)0, acc[s], false, false);

        aC = aN;
        #pragma unroll
        for (int s = 0; s < 4; ++s) bC[s] = bN[s];
    }

    const int rbase = r0 + ((lane < 16) ? 0 : 8);
    #pragma unroll
    for (int s = 0; s < 4; ++s) {
        const int col = c0 + s * 16 + (lane & 15);
        const float bl = b_lin[col];
        #pragma unroll
        for (int r = 0; r < 8; ++r)
            xl[(size_t)(rbase + r) * NN + col] = acc[s][r] + bl;
    }
}

// ---------------- K3: scores = softmax(xl @ Wsel[p]^T + bsel) ---------------
// One wave per (p,b): lanes split K, cross-lane tree reduce, lane0 softmax.
__global__ __launch_bounds__(256) void scores_kernel(const float* __restrict__ xl,
                                                     const float* __restrict__ Wsel,
                                                     const float* __restrict__ bsel,
                                                     float* __restrict__ scores) {
    const int wave = (blockIdx.x * blockDim.x + threadIdx.x) >> 5;
    const int lane = threadIdx.x & 31;
    const int p = wave >> 11;          // /BB
    const int b = wave & (BB - 1);

    float acc[MM];
    #pragma unroll
    for (int m = 0; m < MM; ++m) acc[m] = 0.0f;

    for (int k = lane; k < KK; k += 32) {
        const float xv = xl[(size_t)b * NN + k];
        #pragma unroll
        for (int m = 0; m < MM; ++m)
            acc[m] += xv * Wsel[(size_t)(p * MM + m) * NN + k];
    }
    #pragma unroll
    for (int m = 0; m < MM; ++m)
        for (int off = 16; off > 0; off >>= 1)
            acc[m] += __shfl_down(acc[m], off, 32);

    if (lane == 0) {
        float mx = -1e30f;
        #pragma unroll
        for (int m = 0; m < MM; ++m) { acc[m] += bsel[p * MM + m]; mx = fmaxf(mx, acc[m]); }
        float sum = 0.0f, e[MM];
        #pragma unroll
        for (int m = 0; m < MM; ++m) { e[m] = __expf(acc[m] - mx); sum += e[m]; }
        const float inv = 1.0f / sum;
        #pragma unroll
        for (int m = 0; m < MM; ++m)
            scores[((size_t)p * BB + b) * MM + m] = e[m] * inv;
    }
}

// ---------------- K3b: suppression factor per (p,m) -------------------------
__global__ __launch_bounds__(256) void suppf_kernel(const float* __restrict__ scores,
                                                    const float* __restrict__ ms_thr,
                                                    float* __restrict__ suppf) {
    const int wave = (blockIdx.x * blockDim.x + threadIdx.x) >> 5;  // 64 waves
    const int lane = threadIdx.x & 31;
    const int p = wave >> 3, m = wave & 7;
    float s = 0.0f;
    for (int b = lane; b < BB; b += 32)
        s += scores[((size_t)p * BB + b) * MM + m];
    for (int off = 16; off > 0; off >>= 1) s += __shfl_down(s, off, 32);
    if (lane == 0) {
        const float avg = s * (1.0f / (float)BB);
        suppf[p * MM + m] = (avg > ms_thr[p]) ? 0.1f : 1.0f;
    }
}

// ---------------- K4: per-(p,b) spike reduction -> means, s_pb --------------
__global__ __launch_bounds__(256) void dyn_reduce_kernel(const float* __restrict__ xl,
                                                         const float* __restrict__ scores,
                                                         const float* __restrict__ suppf,
                                                         const float* __restrict__ Wf,
                                                         const float* __restrict__ spring_k,
                                                         float* __restrict__ means,
                                                         float* __restrict__ spb_out) {
    const int wave = (blockIdx.x * blockDim.x + threadIdx.x) >> 5;
    const int lane = threadIdx.x & 31;
    const int p = wave >> 11;
    const int b = wave & (BB - 1);

    float sc[MM], spb = 0.0f;
    #pragma unroll
    for (int m = 0; m < MM; ++m) {
        sc[m] = scores[((size_t)p * BB + b) * MM + m];
        spb += sc[m] * suppf[p * MM + m];
    }
    const float kc = 0.5f * spring_k[p] * (0.01f * 0.01f);

    float cnt = 0.0f;
    for (int n = lane; n < NN; n += 32) {
        const float xv = xl[(size_t)b * NN + n];
        float eq = 0.0f;
        #pragma unroll
        for (int m = 0; m < MM; ++m)
            eq += sc[m] * Wf[(size_t)(p * MM + m) * NN + n];
        const float nx = xv - kc * (xv - eq);
        cnt += (nx > 1.0f) ? 1.0f : 0.0f;
    }
    for (int off = 16; off > 0; off >>= 1) cnt += __shfl_down(cnt, off, 32);
    if (lane == 0) {
        means[(size_t)p * BB + b]   = spb * (cnt * (1.0f / (float)NN));
        spb_out[(size_t)p * BB + b] = spb;
    }
}

// ---------------- K4b: coupling forces + global inhibition (1 block) --------
__global__ __launch_bounds__(256) void coupling_kernel(const float* __restrict__ means,
                                                       const float* __restrict__ coupling,
                                                       const float* __restrict__ g_inh,
                                                       float* __restrict__ cfsum,
                                                       float* __restrict__ inh_out) {
    __shared__ float red[256];
    const int tid = threadIdx.x;
    float local = 0.0f;
    for (int b = tid; b < BB; b += 256) {
        float cs = 0.0f;
        #pragma unroll
        for (int p = 0; p < PP; ++p) {
            float cf = 0.0f;
            #pragma unroll
            for (int q = 0; q < PP; ++q)
                cf += coupling[p * PP + q] * means[(size_t)q * BB + b];
            cs += cf;
            local += means[(size_t)p * BB + b];
        }
        cfsum[b] = cs;
    }
    red[tid] = local;
    __syncthreads();
    for (int s = 128; s > 0; s >>= 1) {
        if (tid < s) red[tid] += red[tid + s];
        __syncthreads();
    }
    if (tid == 0)
        inh_out[0] = g_inh[0] * (red[0] * (1.0f / (float)(PP * BB)));
}

// ---------------- K5: fused epilogue + 537MB poly_mems stream ---------------
// Block handles 8 batch rows x 1024 cols.  Per-block scalars staged in LDS.
// poly_mems stores use NT temporal hint so the stream does not evict the
// L2-resident operands (Wf, xl, scores).
__global__ __launch_bounds__(256) void final_kernel(const float* __restrict__ xl,
                                                    const float* __restrict__ scores,
                                                    const float* __restrict__ spb,
                                                    const float* __restrict__ Wf,
                                                    const float* __restrict__ spring_k,
                                                    const float* __restrict__ cfsum,
                                                    const float* __restrict__ inh_p,
                                                    float* __restrict__ out_final,
                                                    float* __restrict__ out_regmem,
                                                    float* __restrict__ out_poly) {
    __shared__ float s_sc[8][PP][MM];   // [bb][p][m]
    __shared__ float s_spb[8][PP];      // [bb][p]
    __shared__ float s_cf[8];
    __shared__ float s_kc[PP];
    const int tid = threadIdx.x;
    const int b0 = blockIdx.x * 8;

    for (int i = tid; i < 8 * PP * MM; i += 256) {
        const int bb = i >> 6, p = (i >> 3) & 7, m = i & 7;
        s_sc[bb][p][m] = scores[((size_t)p * BB + (b0 + bb)) * MM + m];
    }
    for (int i = tid; i < 8 * PP; i += 256) {
        const int bb = i >> 3, p = i & 7;
        s_spb[bb][p] = spb[(size_t)p * BB + (b0 + bb)];
    }
    if (tid < 8)                 s_cf[tid] = cfsum[b0 + tid] * (1.0f / (float)PP);
    if (tid >= 8 && tid < 16)    s_kc[tid - 8] = 0.5f * spring_k[tid - 8] * (0.01f * 0.01f);
    __syncthreads();

    const float inh = inh_p[0];
    for (int bb = 0; bb < 8; ++bb) {
        const int b = b0 + bb;
        #pragma unroll
        for (int i = 0; i < 4; ++i) {
            const int n = i * 256 + tid;
            const float xv = xl[(size_t)b * NN + n];
            float acc = 0.0f;
            #pragma unroll
            for (int p = 0; p < PP; ++p) {
                float eq = 0.0f;
                #pragma unroll
                for (int m = 0; m < MM; ++m)
                    eq += s_sc[bb][p][m] * Wf[(size_t)(p * MM + m) * NN + n];
                const float nx  = xv - s_kc[p] * (xv - eq);
                const float spk = (nx > 1.0f) ? 1.0f : 0.0f;
                acc += spk * s_spb[bb][p];
                #pragma unroll
                for (int m = 0; m < MM; ++m)
                    __builtin_nontemporal_store(
                        nx, &out_poly[(((size_t)(p * MM + m) * BB + b) * NN) + n]);
            }
            const float reg = (xv > 1.0f) ? 1.0f : 0.0f;
            out_final[(size_t)b * NN + n]  = reg + acc * (1.0f / (float)PP) + s_cf[bb] - inh;
            out_regmem[(size_t)b * NN + n] = xv;
        }
    }
}

// ---------------------------------------------------------------------------
extern "C" void kernel_launch(void* const* d_in, const int* in_sizes, int n_in,
                              void* d_out, int out_size, void* d_ws, size_t ws_size,
                              hipStream_t stream) {
    const float* x        = (const float*)d_in[0];
    const float* W_lin    = (const float*)d_in[1];
    const float* b_lin    = (const float*)d_in[2];
    const float* Wsel     = (const float*)d_in[3];
    const float* bsel     = (const float*)d_in[4];
    const float* Wf       = (const float*)d_in[5];
    const float* spring_k = (const float*)d_in[6];
    // d_in[7] damping_c unused by the reference math (velocity == 0 first step)
    const float* ms_thr   = (const float*)d_in[8];
    const float* coupling = (const float*)d_in[9];
    const float* g_inh    = (const float*)d_in[10];

    char* ws = (char*)d_ws;
    size_t off = 0;
    auto carve = [&](size_t bytes) { char* p = ws + off; off = (off + bytes + 255) & ~(size_t)255; return p; };
    unsigned short* xb = (unsigned short*)carve((size_t)BB * KK * 2);   // 4 MB
    unsigned short* wb = (unsigned short*)carve((size_t)NN * KK * 2);   // 2 MB
    float* xl      = (float*)carve((size_t)BB * NN * 4);                // 8 MB
    float* scores  = (float*)carve((size_t)PP * BB * MM * 4);           // 512 KB
    float* suppf   = (float*)carve((size_t)PP * MM * 4);
    float* means   = (float*)carve((size_t)PP * BB * 4);
    float* spb     = (float*)carve((size_t)PP * BB * 4);
    float* cfsum   = (float*)carve((size_t)BB * 4);
    float* inh     = (float*)carve(4);

    float* out        = (float*)d_out;
    float* out_final  = out;
    float* out_regmem = out + (size_t)BB * NN;
    float* out_poly   = out + (size_t)2 * BB * NN;

    // K1: convert GEMM operands to bf16 (RNE)
    cvt_bf16_kernel<<<(BB * KK + 255) / 256, 256, 0, stream>>>(x, xb, BB * KK);
    cvt_bf16_kernel<<<(NN * KK + 255) / 256, 256, 0, stream>>>(W_lin, wb, NN * KK);

    // K2: xl = x @ W_lin^T + b_lin  (v_wmma_f32_16x16x32_bf16, double-buffered)
    // 2048 waves = 128 row-tiles x 16 col-strips; 8 waves / block
    gemm_xl_kernel<<<(128 * 16) / 8, 256, 0, stream>>>(xb, wb, b_lin, xl);

    // K3: mode routing softmax (wave per (p,b))
    scores_kernel<<<(PP * BB * 32) / 256, 256, 0, stream>>>(xl, Wsel, bsel, scores);

    // K3b: microsleep suppression factors (64 waves)
    suppf_kernel<<<(PP * MM * 32) / 256, 256, 0, stream>>>(scores, ms_thr, suppf);

    // K4: spike-fraction reductions -> means, s_pb (wave per (p,b))
    dyn_reduce_kernel<<<(PP * BB * 32) / 256, 256, 0, stream>>>(xl, scores, suppf, Wf,
                                                                spring_k, means, spb);

    // K4b: coupling @ means + global inhibition (deterministic single block)
    coupling_kernel<<<1, 256, 0, stream>>>(means, coupling, g_inh, cfsum, inh);

    // K5: fused output epilogue + poly_mems broadcast stream
    final_kernel<<<BB / 8, 256, 0, stream>>>(xl, scores, spb, Wf, spring_k, cfsum, inh,
                                             out_final, out_regmem, out_poly);
}
